// MultiheadAttention_8821862826782
// MI455X (gfx1250) — matrix operand order repaired and checked by reading, not compiled, not run on hardware
//
#include <hip/hip_runtime.h>

typedef __attribute__((ext_vector_type(16))) __bf16        v16bf;
typedef __attribute__((ext_vector_type(8)))  float         v8f;
typedef __attribute__((ext_vector_type(4)))  unsigned int  v4u;

union FragU {
    v16bf v;
    v4u   u[2];
};

__device__ __forceinline__ unsigned short f2bf(float f) {
    unsigned int u = __float_as_uint(f);
    u += 0x7FFFu + ((u >> 16) & 1u);   // round to nearest even
    return (unsigned short)(u >> 16);
}

__device__ __forceinline__ v8f wmma_bf16(const FragU& a, const FragU& b, v8f c) {
    return __builtin_amdgcn_wmma_f32_16x16x32_bf16(
        false, a.v, false, b.v, (short)0, c, false, false);
}

__device__ __forceinline__ float rmax16(float v) {   // reduce within 16-lane half-wave
    v = fmaxf(v, __shfl_xor(v, 1, 32));
    v = fmaxf(v, __shfl_xor(v, 2, 32));
    v = fmaxf(v, __shfl_xor(v, 4, 32));
    v = fmaxf(v, __shfl_xor(v, 8, 32));
    return v;
}
__device__ __forceinline__ float rsum16(float v) {
    v += __shfl_xor(v, 1, 32);
    v += __shfl_xor(v, 2, 32);
    v += __shfl_xor(v, 4, 32);
    v += __shfl_xor(v, 8, 32);
    return v;
}

// ---------------------------------------------------------------------------
// f32 -> bf16 bulk convert (4 elements / thread, packed stores)
// ---------------------------------------------------------------------------
__global__ void __launch_bounds__(256)
f32_to_bf16_kernel(const float* __restrict__ s, unsigned short* __restrict__ d, int n) {
    int i = (blockIdx.x * 256 + threadIdx.x) * 4;
    if (i >= n) return;
    float4 f = *(const float4*)(s + i);
    unsigned int lo = (unsigned int)f2bf(f.x) | ((unsigned int)f2bf(f.y) << 16);
    unsigned int hi = (unsigned int)f2bf(f.z) | ((unsigned int)f2bf(f.w) << 16);
    *(uint2*)(d + i) = make_uint2(lo, hi);
}

// ---------------------------------------------------------------------------
// WMMA GEMM for the 4096x1024x1024 projections: C = scale*(A@W + bias)
//   OUT_MODE 0: f32 row-major [M][N]                   (output projection)
//   OUT_MODE 2: bf16 head-split [B,H,S,64]             (Q, K projections)
//   OUT_MODE 3: bf16 transposed head-split [B,H,64,S]  (V projection)
// A bf16 [M][K] row-major; W bf16 [K][N] row-major (transposed while staged).
// Block = 256 threads (8 waves), tile 128(M) x 64(N), K-step 32.
// ---------------------------------------------------------------------------
template <int OUT_MODE>
__global__ void __launch_bounds__(256)
gemm_wmma(const unsigned short* __restrict__ A, const unsigned short* __restrict__ W,
          const float* __restrict__ bias, void* __restrict__ C,
          int M, int N, int K, float scale) {
    __shared__ unsigned short lA[128][40];   // 80B pitch: 16B-aligned, conflict-spread
    __shared__ unsigned short lB[64][40];

    const int tid   = threadIdx.x;
    const int wv    = tid >> 5;
    const int lane  = tid & 31;
    const int mBase = blockIdx.y * 128;
    const int nBase = blockIdx.x * 64;

    v8f acc[4] = {};

    const int laneN = lane & 15;
    const int half  = lane >> 4;
    const int r     = wv * 16 + laneN;   // A fragment row in LDS tile
    const int kbA   = half * 16;         // A half-wave contiguous K {0-15}/{16-31}
    const int kbB   = half * 16;         // B half-wave contiguous K {0-15}/{16-31}

    for (int k0 = 0; k0 < K; k0 += 32) {
        #pragma unroll
        for (int i = 0; i < 2; ++i) {                 // A tile 128x32, b128 copies
            int task  = tid + i * 256;
            int row   = task >> 2;
            int chunk = task & 3;
            v4u d = *(const v4u*)(A + (size_t)(mBase + row) * K + k0 + chunk * 8);
            *(v4u*)&lA[row][chunk * 8] = d;
        }
        #pragma unroll
        for (int i = 0; i < 8; ++i) {                 // W tile 32x64, transpose to [n][k]
            int task = tid + i * 256;
            int k = task >> 6, n = task & 63;
            lB[n][k] = W[(size_t)(k0 + k) * N + nBase + n];
        }
        __syncthreads();

        FragU a;
        a.u[0] = *(const v4u*)&lA[r][kbA];
        a.u[1] = *(const v4u*)&lA[r][kbA + 8];
        #pragma unroll
        for (int c = 0; c < 4; ++c) {
            FragU b;
            b.u[0] = *(const v4u*)&lB[c * 16 + laneN][kbB];
            b.u[1] = *(const v4u*)&lB[c * 16 + laneN][kbB + 8];
            acc[c] = wmma_bf16(a, b, acc[c]);
        }
        __syncthreads();
    }

    #pragma unroll
    for (int c = 0; c < 4; ++c) {
        int coln = nBase + c * 16 + laneN;
        float bb = bias ? bias[coln] : 0.0f;
        #pragma unroll
        for (int j = 0; j < 8; ++j) {
            int   m = mBase + wv * 16 + j + half * 8;
            float v = (acc[c][j] + bb) * scale;
            if (OUT_MODE == 0) {
                ((float*)C)[(size_t)m * N + coln] = v;
            } else if (OUT_MODE == 2) {   // [B,H,S,64]
                size_t idx = ((size_t)((m >> 11) * 16 + (coln >> 6))) * (2048 * 64)
                           + (size_t)(m & 2047) * 64 + (coln & 63);
                ((unsigned short*)C)[idx] = f2bf(v);
            } else {                      // [B,H,64,S]
                size_t idx = ((size_t)((m >> 11) * 16 + (coln >> 6))) * (64 * 2048)
                           + (size_t)(coln & 63) * 2048 + (m & 2047);
                ((unsigned short*)C)[idx] = f2bf(v);
            }
        }
    }
}

// ---------------------------------------------------------------------------
// Fused flash attention; one wave owns 16 query rows, no inter-wave barriers.
//   Qh [BH][S][64] bf16 (pre-scaled by 1/sqrt(d_model))
//   Kh [BH][S][64] bf16
//   Vt [BH][64][S] bf16 (transposed -> P*V B-fragments are k-contiguous)
// Exact softmax in three S sweeps (K tiles are L2-resident, so recomputing
// S with WMMA is cheaper than per-block ds_bpermute reductions):
//   A: per-lane running max  (no shuffles in loop; 1 reduce at end)
//   B: per-lane running sum of exp(S-m)   (1 reduce at end)
//   C: P = exp(S-m)/l -> write attn once, LDS relayout, O = P @ V
// ---------------------------------------------------------------------------
__global__ void __launch_bounds__(256)
flash_attn(const unsigned short* __restrict__ Qh, const unsigned short* __restrict__ Kh,
           const unsigned short* __restrict__ Vt, float* __restrict__ attn,
           unsigned short* __restrict__ Xbf) {
    __shared__ unsigned short lp[8][16][40];   // per-wave P staging (80B pitch)

    const int tid   = threadIdx.x;
    const int wv    = tid >> 5;
    const int lane  = tid & 31;
    const int laneN = lane & 15;
    const int half  = lane >> 4;
    const int kbA   = half * 16;
    const int kbB   = half * 16;

    const long long bh    = blockIdx.z;
    const int       qBase = blockIdx.x * 128 + wv * 16;   // local q row of this wave

    const unsigned short* Q = Qh + bh * (size_t)(2048 * 64);
    const unsigned short* K = Kh + bh * (size_t)(2048 * 64);
    const unsigned short* V = Vt + bh * (size_t)(64 * 2048);
    float* attnP = attn + bh * (2048LL * 2048);

    // Q fragments for this wave's 16 rows (d split 0-31 / 32-63), kept in VGPRs
    const unsigned short* qrow = Q + (size_t)(qBase + laneN) * 64;
    FragU qf[2];
    qf[0].u[0] = *(const v4u*)(qrow + kbA);
    qf[0].u[1] = *(const v4u*)(qrow + kbA + 8);
    qf[1].u[0] = *(const v4u*)(qrow + 32 + kbA);
    qf[1].u[1] = *(const v4u*)(qrow + 32 + kbA + 8);

    // ---------------- pass A: exact row max ----------------
    float rowM[8];
    #pragma unroll
    for (int j = 0; j < 8; ++j) rowM[j] = -3.0e38f;
    for (int kb = 0; kb < 2048; kb += 32) {
        v8f s[2] = {};
        #pragma unroll
        for (int t = 0; t < 2; ++t) {
            const unsigned short* krow = K + (size_t)(kb + t * 16 + laneN) * 64;
            FragU b0, b1;
            b0.u[0] = *(const v4u*)(krow + kbB);      b0.u[1] = *(const v4u*)(krow + kbB + 8);
            b1.u[0] = *(const v4u*)(krow + 32 + kbB); b1.u[1] = *(const v4u*)(krow + 32 + kbB + 8);
            s[t] = wmma_bf16(qf[0], b0, s[t]);
            s[t] = wmma_bf16(qf[1], b1, s[t]);
        }
        #pragma unroll
        for (int j = 0; j < 8; ++j)
            rowM[j] = fmaxf(rowM[j], fmaxf(s[0][j], s[1][j]));
    }
    #pragma unroll
    for (int j = 0; j < 8; ++j) rowM[j] = rmax16(rowM[j]);

    // ---------------- pass B: exact row sum of exp(S - m) ----------------
    float rowL[8];
    #pragma unroll
    for (int j = 0; j < 8; ++j) rowL[j] = 0.0f;
    for (int kb = 0; kb < 2048; kb += 32) {
        v8f s[2] = {};
        #pragma unroll
        for (int t = 0; t < 2; ++t) {
            const unsigned short* krow = K + (size_t)(kb + t * 16 + laneN) * 64;
            FragU b0, b1;
            b0.u[0] = *(const v4u*)(krow + kbB);      b0.u[1] = *(const v4u*)(krow + kbB + 8);
            b1.u[0] = *(const v4u*)(krow + 32 + kbB); b1.u[1] = *(const v4u*)(krow + 32 + kbB + 8);
            s[t] = wmma_bf16(qf[0], b0, s[t]);
            s[t] = wmma_bf16(qf[1], b1, s[t]);
        }
        #pragma unroll
        for (int j = 0; j < 8; ++j)
            rowL[j] += __expf(s[0][j] - rowM[j]) + __expf(s[1][j] - rowM[j]);
    }
    float invL[8];
    #pragma unroll
    for (int j = 0; j < 8; ++j) invL[j] = 1.0f / rsum16(rowL[j]);

    // ---------------- pass C: write P once, O = P @ V ----------------
    v8f o[4] = {};
    for (int kb = 0; kb < 2048; kb += 32) {
        v8f s[2] = {};
        #pragma unroll
        for (int t = 0; t < 2; ++t) {
            const unsigned short* krow = K + (size_t)(kb + t * 16 + laneN) * 64;
            FragU b0, b1;
            b0.u[0] = *(const v4u*)(krow + kbB);      b0.u[1] = *(const v4u*)(krow + kbB + 8);
            b1.u[0] = *(const v4u*)(krow + 32 + kbB); b1.u[1] = *(const v4u*)(krow + 32 + kbB + 8);
            s[t] = wmma_bf16(qf[0], b0, s[t]);
            s[t] = wmma_bf16(qf[1], b1, s[t]);
        }
        // P = exp(S - m) / l : store to d_out (only attn write) + LDS relayout
        #pragma unroll
        for (int t = 0; t < 2; ++t) {
            #pragma unroll
            for (int j = 0; j < 8; ++j) {
                float p = __expf(s[t][j] - rowM[j]) * invL[j];
                attnP[(size_t)(qBase + j + 8 * half) * 2048 + kb + t * 16 + laneN] = p;
                lp[wv][j + 8 * half][t * 16 + laneN] = f2bf(p);
            }
        }
        // P as 16x32 A-fragment (cross-lane relayout via this wave's LDS slab)
        FragU pf;
        pf.u[0] = *(const v4u*)&lp[wv][laneN][kbA];
        pf.u[1] = *(const v4u*)&lp[wv][laneN][kbA + 8];
        #pragma unroll
        for (int c = 0; c < 4; ++c) {
            const unsigned short* vrow = V + (size_t)(c * 16 + laneN) * 2048 + kb;
            FragU b;
            b.u[0] = *(const v4u*)(vrow + kbB);
            b.u[1] = *(const v4u*)(vrow + kbB + 8);
            o[c] = wmma_bf16(pf, b, o[c]);
        }
    }

    // context -> Xbf [B][S][1024] bf16 (already normalized by invL)
    size_t xbase = (size_t)(bh >> 4) * (2048 * 1024) + (size_t)(bh & 15) * 64;
    #pragma unroll
    for (int c = 0; c < 4; ++c) {
        #pragma unroll
        for (int j = 0; j < 8; ++j) {
            Xbf[xbase + (size_t)(qBase + j + 8 * half) * 1024 + c * 16 + laneN] =
                f2bf(o[c][j]);
        }
    }
}

// ---------------------------------------------------------------------------
extern "C" void kernel_launch(void* const* d_in, const int* in_sizes, int n_in,
                              void* d_out, int out_size, void* d_ws, size_t ws_size,
                              hipStream_t stream) {
    constexpr int B = 2, S = 2048, D = 1024, H = 16;
    constexpr int BSD = B * S * D;       // 4,194,304
    constexpr int DD  = D * D;           // 1,048,576

    const float* query = (const float*)d_in[0];
    const float* key   = (const float*)d_in[1];
    const float* value = (const float*)d_in[2];
    const float* Wq    = (const float*)d_in[3];
    const float* bq    = (const float*)d_in[4];
    const float* Wk    = (const float*)d_in[5];
    const float* bk    = (const float*)d_in[6];
    const float* Wv    = (const float*)d_in[7];
    const float* bv    = (const float*)d_in[8];
    const float* Wo    = (const float*)d_in[9];
    const float* bo    = (const float*)d_in[10];

    // d_out = [out (B,S,D) f32][attn (B,H,S,S) f32]
    float* outp = (float*)d_out;
    float* attn = outp + (size_t)BSD;

    // workspace (bf16 shorts)
    unsigned short* w = (unsigned short*)d_ws;
    unsigned short* qbf = w;  w += BSD;
    unsigned short* kbf = w;  w += BSD;
    unsigned short* vbf = w;  w += BSD;
    unsigned short* wqb = w;  w += DD;
    unsigned short* wkb = w;  w += DD;
    unsigned short* wvb = w;  w += DD;
    unsigned short* wob = w;  w += DD;
    unsigned short* Qh  = w;  w += BSD;   // [B,H,S,64]
    unsigned short* Kh  = w;  w += BSD;   // [B,H,S,64]
    unsigned short* Vt  = w;  w += BSD;   // [B,H,64,S]
    unsigned short* Xbf = w;  w += BSD;   // [B,S,D]

    // 1) f32 -> bf16
    f32_to_bf16_kernel<<<BSD / 1024, 256, 0, stream>>>(query, qbf, BSD);
    f32_to_bf16_kernel<<<BSD / 1024, 256, 0, stream>>>(key,   kbf, BSD);
    f32_to_bf16_kernel<<<BSD / 1024, 256, 0, stream>>>(value, vbf, BSD);
    f32_to_bf16_kernel<<<DD  / 1024, 256, 0, stream>>>(Wq,    wqb, DD);
    f32_to_bf16_kernel<<<DD  / 1024, 256, 0, stream>>>(Wk,    wkb, DD);
    f32_to_bf16_kernel<<<DD  / 1024, 256, 0, stream>>>(Wv,    wvb, DD);
    f32_to_bf16_kernel<<<DD  / 1024, 256, 0, stream>>>(Wo,    wob, DD);

    // 2) projections (1/sqrt(d_model) folded into Q); V stored transposed
    {
        dim3 g(D / 64, (B * S) / 128, 1);
        gemm_wmma<2><<<g, 256, 0, stream>>>(qbf, wqb, bq, Qh, B * S, D, D, 0.03125f);
        gemm_wmma<2><<<g, 256, 0, stream>>>(kbf, wkb, bk, Kh, B * S, D, D, 1.0f);
        gemm_wmma<3><<<g, 256, 0, stream>>>(vbf, wvb, bv, Vt, B * S, D, D, 1.0f);
    }

    // 3) fused attention: scores + softmax + attn write + context
    {
        dim3 g(S / 128, 1, B * H);
        flash_attn<<<g, 256, 0, stream>>>(Qh, Kh, Vt, attn, Xbf);
    }

    // 4) out = context @ Wo + bo
    {
        dim3 g(D / 64, (B * S) / 128, 1);
        gemm_wmma<0><<<g, 256, 0, stream>>>(Xbf, wob, bo, outp, B * S, D, D, 1.0f);
    }
    (void)in_sizes; (void)n_in; (void)out_size; (void)ws_size;
}
